// SingleAttention_10780367913459
// MI455X (gfx1250) — compile-verified, matrix operand order checked by reading, not run on hardware
//
#include <hip/hip_runtime.h>

typedef __attribute__((ext_vector_type(16))) _Float16 v16h;
typedef __attribute__((ext_vector_type(8)))  float    v8f;
typedef __attribute__((ext_vector_type(4)))  float    v4f;   // native vector for NT stores

// Problem sizes from the reference (fixed).
constexpr int B = 64, C = 512, H = 32, W = 32;
constexpr int THREADS    = 256;             // 8 wave32s per workgroup
constexpr int WAVES      = THREADS / 32;    // 8
constexpr int C_PER_WAVE = C / WAVES;       // 64 channels per wave
constexpr int TILES      = C_PER_WAVE / 16; // 4 WMMA tiles per wave

__global__ __launch_bounds__(THREADS)
void single_attention_fused(const float* __restrict__ F,
                            const float* __restrict__ v0,
                            float* __restrict__ out)
{
    __shared__ float s_v0[C];
    __shared__ float s_partial[WAVES];
    __shared__ float s_att;

    const int bh   = blockIdx.x;      // one (b,h) pair per workgroup
    const int b    = bh / H;
    const int h    = bh % H;
    const int tid  = threadIdx.x;
    const int wave = tid >> 5;
    const int lane = tid & 31;
    const int half = lane >> 4;       // which half-wave (A-matrix K split)
    const int m    = lane & 15;       // row within the 16-row tile

    // Stage v0 (2 KB) into LDS once.
    for (int i = tid; i < C; i += THREADS) s_v0[i] = v0[i];
    __syncthreads();

    // B-matrix of ones: turns WMMA into a per-row sum reduction over K=32.
    v16h bones;
    #pragma unroll
    for (int i = 0; i < 16; ++i) bones[i] = (_Float16)1.0f;

    // A-matrix 16x32 f16 lane layout (ISA 7.12.2):
    //   lanes 0-15  : row M=lane,    K = 0..7   (VGPR0-3) and 16..23 (VGPR4-7)
    //   lanes 16-31 : row M=lane-16, K = 8..15  (VGPR0-3) and 24..31 (VGPR4-7)
    const int w0 = 8 * half;        // first contiguous 8-element K group
    const int w1 = 16 + 8 * half;   // second contiguous 8-element K group

    float att = 0.0f;

    #pragma unroll
    for (int t = 0; t < TILES; ++t) {
        const int c = wave * C_PER_WAVE + t * 16 + m;   // this lane's A-row channel
        const float4* row =
            (const float4*)(F + (((size_t)b * C + c) * H + h) * W);

        const float4 r0 = row[(w0 >> 2) + 0];
        const float4 r1 = row[(w0 >> 2) + 1];
        const float4 r2 = row[(w1 >> 2) + 0];
        const float4 r3 = row[(w1 >> 2) + 1];

        v16h a;
        a[0]  = (_Float16)(r0.x * r0.x);
        a[1]  = (_Float16)(r0.y * r0.y);
        a[2]  = (_Float16)(r0.z * r0.z);
        a[3]  = (_Float16)(r0.w * r0.w);
        a[4]  = (_Float16)(r1.x * r1.x);
        a[5]  = (_Float16)(r1.y * r1.y);
        a[6]  = (_Float16)(r1.z * r1.z);
        a[7]  = (_Float16)(r1.w * r1.w);
        a[8]  = (_Float16)(r2.x * r2.x);
        a[9]  = (_Float16)(r2.y * r2.y);
        a[10] = (_Float16)(r2.z * r2.z);
        a[11] = (_Float16)(r2.w * r2.w);
        a[12] = (_Float16)(r3.x * r3.x);
        a[13] = (_Float16)(r3.y * r3.y);
        a[14] = (_Float16)(r3.z * r3.z);
        a[15] = (_Float16)(r3.w * r3.w);

        v8f acc = {};
        // D[m][n] = sum_k A[m][k] * 1 = sum_w F^2  (replicated across n)
        acc = __builtin_amdgcn_wmma_f32_16x16x32_f16(
            /*neg_a=*/false, a, /*neg_b=*/false, bones,
            /*c_mod=*/(short)0, acc, /*reuse_a=*/false, /*reuse_b=*/false);

        // D layout: VGPR r -> row r (lanes 0-15), row r+8 (lanes 16-31).
        const int cbase = wave * C_PER_WAVE + t * 16 + 8 * half;
        #pragma unroll
        for (int r = 0; r < 8; ++r)
            att += s_v0[cbase + r] * __builtin_amdgcn_sqrtf(acc[r]);
    }

    // Combine the two half-wave partials (rows 0-7 vs 8-15 of each tile).
    att += __shfl_xor(att, 16, 32);
    if (lane == 0) s_partial[wave] = att;
    __syncthreads();

    if (tid == 0) {
        float s = 0.0f;
        #pragma unroll
        for (int i = 0; i < WAVES; ++i) s += s_partial[i];
        s_att = s;   // att[b,h]
    }
    __syncthreads();
    const float s = s_att;

    // Phase 2: out[b,:,h,:] = s * F[b,:,h,:]. The slab (64 KB) was just read,
    // so these loads hit L2. Output is write-once/never-read -> non-temporal
    // stores, so the 128 MB of output doesn't evict F_r (128 MB) from the
    // 192 MB L2 and the phase-2 re-read stays on-chip.
    constexpr int SLAB4 = C * W / 4;   // 4096 float4 per (b,h)
    #pragma unroll 4
    for (int j = tid; j < SLAB4; j += THREADS) {
        const int c  = j >> 3;         // j / (W/4)
        const int w4 = j & 7;
        const size_t g4 = (((size_t)b * C + c) * H + h) * (W / 4) + w4;
        const v4f fv = ((const v4f*)F)[g4];
        const v4f ov = s * fv;
        __builtin_nontemporal_store(ov, ((v4f*)out) + g4);
    }
}

extern "C" void kernel_launch(void* const* d_in, const int* in_sizes, int n_in,
                              void* d_out, int out_size, void* d_ws, size_t ws_size,
                              hipStream_t stream)
{
    (void)in_sizes; (void)n_in; (void)out_size; (void)d_ws; (void)ws_size;
    const float* F  = (const float*)d_in[0];   // F_r  [B,C,H,W] f32
    const float* v0 = (const float*)d_in[1];   // v0   [1,C,1,1] f32
    float* out      = (float*)d_out;           // out  [B,C,H,W] f32

    single_attention_fused<<<B * H, THREADS, 0, stream>>>(F, v0, out);
}
